// MultiHeadAttentionBlock_65223373357373
// MI455X (gfx1250) — matrix operand order repaired and checked by reading, not compiled, not run on hardware
//
#include <hip/hip_runtime.h>

typedef __attribute__((ext_vector_type(16))) _Float16 v16h;
typedef __attribute__((ext_vector_type(8)))  float    v8f;

#define B_    8
#define C_    512
#define T_    1024
#define NH_   8
#define HD_   64
#define G_    32
#define CPG_  16
#define EPS_  1e-5f
#define SCALE_ 0.125f   // 1/sqrt(64)

// ---------------- WMMA helpers (gfx1250: 16x16x32 f16 -> f32) ----------------

static __device__ __forceinline__ v8f wmma16x16x32(v16h a, v16h b, v8f c) {
  return __builtin_amdgcn_wmma_f32_16x16x32_f16(false, a, false, b, (short)0, c,
                                                false, false);
}

// A fragment (16x32 f16). ISA layout: lane L -> row m = L&15;
// element e -> k = (L>>4)*8 + (e&7) + (e>>3)*16.  Memory: row-major, contiguous k.
static __device__ __forceinline__ v16h load_a_frag(const _Float16* p, int ld) {
  const int lane = threadIdx.x & 31;
  const _Float16* row = p + (size_t)(lane & 15) * ld + ((lane >> 4) << 3);
  v16h a;
#pragma unroll
  for (int e = 0; e < 8; ++e) a[e] = row[e];
#pragma unroll
  for (int e = 0; e < 8; ++e) a[e + 8] = row[e + 16];
  return a;
}

// B fragment (32x16 f16). ISA layout: lane L -> col n = L&15;
// element e -> k = (L>>4)*8 + (e&7) + (e>>3)*16.  Memory: mem[n][k] row-major, contiguous k.
static __device__ __forceinline__ v16h load_b_frag(const _Float16* p, int ld) {
  const int lane = threadIdx.x & 31;
  const _Float16* row = p + (size_t)(lane & 15) * ld + ((lane >> 4) << 3);
  v16h b;
#pragma unroll
  for (int e = 0; e < 8; ++e) b[e] = row[e];
#pragma unroll
  for (int e = 0; e < 8; ++e) b[e + 8] = row[e + 16];
  return b;
}

// ---- CDNA5 async global->LDS copy of one 64x32 f16 B tile (4KB, 16B/thread) --
// Tracked with ASYNCcnt (ISA §10 / CDNA5 ISA).

static __device__ __forceinline__ void async_copy_btile(const _Float16* gsrc,
                                                        _Float16* ldst) {
  const int tid  = threadIdx.x;         // 256 threads
  const int trow = tid >> 2;            // 0..63 (t row)
  const int kk   = (tid & 3) << 3;      // 0,8,16,24 (k halves)
  unsigned long long g =
      (unsigned long long)(uintptr_t)(gsrc + (size_t)trow * C_ + kk);
  unsigned l = (unsigned)(uintptr_t)(ldst + trow * 32 + kk);  // LDS byte addr
  asm volatile("global_load_async_to_lds_b128 %0, %1, off"
               :: "v"(l), "v"(g) : "memory");
}

// 8 WMMAs of one K=32 step: 2 A fragments x 4 shared B fragments from LDS.
static __device__ __forceinline__ void wmma_step(const _Float16* cur,
                                                 v16h a0, v16h a1, v8f acc[2][4]) {
  v16h b0 = load_b_frag(cur, 32);
  v16h b1 = load_b_frag(cur + 512, 32);
  v16h b2 = load_b_frag(cur + 1024, 32);
  v16h b3 = load_b_frag(cur + 1536, 32);
  acc[0][0] = wmma16x16x32(a0, b0, acc[0][0]);
  acc[1][0] = wmma16x16x32(a1, b0, acc[1][0]);
  acc[0][1] = wmma16x16x32(a0, b1, acc[0][1]);
  acc[1][1] = wmma16x16x32(a1, b1, acc[1][1]);
  acc[0][2] = wmma16x16x32(a0, b2, acc[0][2]);
  acc[1][2] = wmma16x16x32(a1, b2, acc[1][2]);
  acc[0][3] = wmma16x16x32(a0, b3, acc[0][3]);
  acc[1][3] = wmma16x16x32(a1, b3, acc[1][3]);
}

// Shared GEMM core: block tile 256(M) x 64(N), wave tile 32x64, K = 512 in 16
// steps of 32.  B tile double-buffered in LDS via async DMA; A pipelined in
// registers.  Steady state unrolled x2 so even/odd steps statically own their
// LDS buffer + A register set (no pipeline-rotation moves).
static __device__ __forceinline__ void gemm_core(const _Float16* __restrict__ wgtA,
                                                 const _Float16* __restrict__ bbase,
                                                 _Float16* bsm, v8f acc[2][4]) {
  const _Float16* wgtB = wgtA + (size_t)16 * C_;
  async_copy_btile(bbase, bsm);                         // tile 0 -> buf0
  v16h a0 = load_a_frag(wgtA, C_);
  v16h a1 = load_a_frag(wgtB, C_);
#pragma unroll 1
  for (int s = 0; s < 14; s += 2) {
    // even step s (buf0); stage tile s+1 -> buf1
    async_copy_btile(bbase + (s + 1) * 32, bsm + 2048);
    v16h a0n = load_a_frag(wgtA + (s + 1) * 32, C_);
    v16h a1n = load_a_frag(wgtB + (s + 1) * 32, C_);
    asm volatile("s_wait_asynccnt 1" ::: "memory");
    __syncthreads();
    wmma_step(bsm, a0, a1, acc);
    __syncthreads();
    // odd step s+1 (buf1); stage tile s+2 -> buf0
    async_copy_btile(bbase + (s + 2) * 32, bsm);
    a0 = load_a_frag(wgtA + (s + 2) * 32, C_);
    a1 = load_a_frag(wgtB + (s + 2) * 32, C_);
    asm volatile("s_wait_asynccnt 1" ::: "memory");
    __syncthreads();
    wmma_step(bsm + 2048, a0n, a1n, acc);
    __syncthreads();
  }
  // step 14 (buf0); stage tile 15 -> buf1
  async_copy_btile(bbase + 15 * 32, bsm + 2048);
  v16h a0n = load_a_frag(wgtA + 15 * 32, C_);
  v16h a1n = load_a_frag(wgtB + 15 * 32, C_);
  asm volatile("s_wait_asynccnt 1" ::: "memory");
  __syncthreads();
  wmma_step(bsm, a0, a1, acc);
  __syncthreads();
  // step 15 (buf1)
  asm volatile("s_wait_asynccnt 0" ::: "memory");
  __syncthreads();
  wmma_step(bsm + 2048, a0n, a1n, acc);
}

// ---------------- Kernel 1: f32 -> f16 weight conversion ----------------

__global__ void cvt_kernel(const float* __restrict__ s, _Float16* __restrict__ d, int n) {
  int i = blockIdx.x * 256 + threadIdx.x;
  if (i < n) d[i] = (_Float16)s[i];
}

// ---------------- Kernel 2: GroupNorm, writes xn t-major [B][T][C] as f16 ------

__global__ void gn_kernel(const float* __restrict__ x, const float* __restrict__ w,
                          const float* __restrict__ bias, _Float16* __restrict__ xnT) {
  const int b = blockIdx.x / G_;
  const int g = blockIdx.x % G_;
  const int tid = threadIdx.x;
  __shared__ float red[256];
  const float* xp = x + ((size_t)b * C_ + g * CPG_) * T_;

  float s = 0.f, ss = 0.f;
  for (int i = tid; i < CPG_ * T_; i += 256) {
    float v = xp[i];
    s += v; ss += v * v;
  }
  red[tid] = s; __syncthreads();
  for (int o = 128; o > 0; o >>= 1) { if (tid < o) red[tid] += red[tid + o]; __syncthreads(); }
  float mean = red[0] / (float)(CPG_ * T_);
  __syncthreads();
  red[tid] = ss; __syncthreads();
  for (int o = 128; o > 0; o >>= 1) { if (tid < o) red[tid] += red[tid + o]; __syncthreads(); }
  float var = red[0] / (float)(CPG_ * T_) - mean * mean;
  float rstd = rsqrtf(var + EPS_);

  for (int i = tid; i < CPG_ * T_; i += 256) {
    int c = g * CPG_ + i / T_;
    int t = i % T_;
    float v = (xp[i] - mean) * rstd * w[c] + bias[c];
    xnT[((size_t)b * T_ + t) * C_ + c] = (_Float16)v;
  }
}

// ---------------- Kernel 3: QKV GEMM  [1536x512] x [512x1024] per batch -------
// Block: 8 waves, tile 256(o) x 64(t); B tile shared in LDS (async DMA).
// Writes qkvT t-major [B][T][3C] and v additionally c-major [B][C][T].

__global__ void qkv_gemm_kernel(const _Float16* __restrict__ wgt,
                                const _Float16* __restrict__ xnT,
                                const float* __restrict__ qkv_b,
                                _Float16* __restrict__ qkvT,
                                _Float16* __restrict__ vbuf) {
  __shared__ _Float16 bsm[2 * 64 * 32];                 // double-buffered B tile
  const int b    = blockIdx.x / 96;
  const int rblk = blockIdx.x % 96;
  const int mt   = rblk >> 4;                           // 0..5
  const int nt   = rblk & 15;                           // 0..15
  const int wave = threadIdx.x >> 5;
  const int lane = threadIdx.x & 31;
  const int m0   = mt * 256 + wave * 32;                // 0..1504
  const int n0   = nt * 64;

  v8f zero = {};
  v8f acc[2][4];
#pragma unroll
  for (int i = 0; i < 2; ++i)
#pragma unroll
    for (int j = 0; j < 4; ++j) acc[i][j] = zero;

  gemm_core(wgt + (size_t)m0 * C_, xnT + ((size_t)b * T_ + n0) * C_, bsm, acc);

  const int hi = lane >> 4;
  const int nl = lane & 15;
#pragma unroll
  for (int i = 0; i < 2; ++i) {
#pragma unroll
    for (int j = 0; j < 4; ++j) {
#pragma unroll
      for (int r = 0; r < 8; ++r) {
        int o = m0 + i * 16 + r + hi * 8;
        int t = n0 + j * 16 + nl;
        _Float16 hv = (_Float16)(acc[i][j][r] + qkv_b[o]);
        qkvT[((size_t)b * T_ + t) * (3 * C_) + o] = hv;
        int which = o % 192;                            // 0..63 q | 64..127 k | 128..191 v
        if (which >= 128)
          vbuf[((size_t)b * C_ + (o / 192) * HD_ + (which - 128)) * T_ + t] = hv;
      }
    }
  }
}

// ---------------- Kernel 4: flash attention per (b,head), 16 q-rows per wave ---

__global__ void attn_kernel(const _Float16* __restrict__ qkvT,
                            const _Float16* __restrict__ vbuf,
                            _Float16* __restrict__ attnT) {
  __shared__ _Float16 plds[4][16 * 32];                 // per-wave P transpose buf
  const int bh   = blockIdx.x >> 4;
  const int b    = bh >> 3, h = bh & 7;
  const int wave = threadIdx.x >> 5;
  const int lane = threadIdx.x & 31;
  const int t0   = ((blockIdx.x & 15) * 4 + wave) << 4;
  const int hi   = lane >> 4;
  const int nl   = lane & 15;

  const _Float16* qbase = qkvT + ((size_t)b * T_ + t0) * (3 * C_) + h * 192;
  v16h aq0 = load_a_frag(qbase, 3 * C_);
  v16h aq1 = load_a_frag(qbase + 32, 3 * C_);

  v8f zero = {};
  v8f O0 = zero, O1 = zero, O2 = zero, O3 = zero;
  float mprev[8], lsum[8];
#pragma unroll
  for (int r = 0; r < 8; ++r) { mprev[r] = -1e30f; lsum[r] = 0.f; }

  _Float16* pl = &plds[wave][0];

  for (int s0 = 0; s0 < T_; s0 += 32) {
    // scores tile 16 x 32 = Q^T(16x64) x K(64x32) : 4 WMMAs
    const _Float16* kb = qkvT + ((size_t)b * T_ + s0) * (3 * C_) + h * 192 + HD_;
    v16h bk00 = load_b_frag(kb, 3 * C_);
    v16h bk01 = load_b_frag(kb + 32, 3 * C_);
    v16h bk10 = load_b_frag(kb + (size_t)16 * (3 * C_), 3 * C_);
    v16h bk11 = load_b_frag(kb + (size_t)16 * (3 * C_) + 32, 3 * C_);
    v8f S0 = wmma16x16x32(aq1, bk01, wmma16x16x32(aq0, bk00, zero));
    v8f S1 = wmma16x16x32(aq1, bk11, wmma16x16x32(aq0, bk10, zero));

    // online softmax; row r+hi*8 lives in the 16 lanes of this half -> xor 1/2/4/8
    float fac[8];
#pragma unroll
    for (int r = 0; r < 8; ++r) {
      float a0 = S0[r] * SCALE_, a1 = S1[r] * SCALE_;
      float mc = fmaxf(a0, a1);
      mc = fmaxf(mc, __shfl_xor(mc, 1, 32));
      mc = fmaxf(mc, __shfl_xor(mc, 2, 32));
      mc = fmaxf(mc, __shfl_xor(mc, 4, 32));
      mc = fmaxf(mc, __shfl_xor(mc, 8, 32));
      float mn = fmaxf(mprev[r], mc);
      fac[r] = __expf(mprev[r] - mn);
      float p0 = __expf(a0 - mn), p1 = __expf(a1 - mn);
      float rs = p0 + p1;
      rs += __shfl_xor(rs, 1, 32);
      rs += __shfl_xor(rs, 2, 32);
      rs += __shfl_xor(rs, 4, 32);
      rs += __shfl_xor(rs, 8, 32);
      lsum[r] = lsum[r] * fac[r] + rs;
      mprev[r] = mn;
      S0[r] = p0; S1[r] = p1;
    }
#pragma unroll
    for (int r = 0; r < 8; ++r) {
      O0[r] *= fac[r]; O1[r] *= fac[r]; O2[r] *= fac[r]; O3[r] *= fac[r];
    }

    // re-layout P (C-fragment) -> A-fragment via LDS
    __syncthreads();
#pragma unroll
    for (int r = 0; r < 8; ++r) {
      int m = r + hi * 8;
      pl[m * 32 + nl]      = (_Float16)S0[r];
      pl[m * 32 + 16 + nl] = (_Float16)S1[r];
    }
    __syncthreads();
    v16h ap = load_a_frag(pl, 32);

    // O(16x64) += P(16x32) x V^T(32x64) : 4 WMMAs; vbuf is c-major -> contiguous
    const _Float16* vb = vbuf + ((size_t)b * C_ + h * HD_) * T_ + s0;
    v16h bv0 = load_b_frag(vb + (size_t)0  * T_, T_);
    v16h bv1 = load_b_frag(vb + (size_t)16 * T_, T_);
    v16h bv2 = load_b_frag(vb + (size_t)32 * T_, T_);
    v16h bv3 = load_b_frag(vb + (size_t)48 * T_, T_);
    O0 = wmma16x16x32(ap, bv0, O0);
    O1 = wmma16x16x32(ap, bv1, O1);
    O2 = wmma16x16x32(ap, bv2, O2);
    O3 = wmma16x16x32(ap, bv3, O3);
  }

  // normalize & store t-major attnT[B][T][C]
#pragma unroll
  for (int r = 0; r < 8; ++r) {
    float inv = 1.f / lsum[r];
    int t = t0 + r + hi * 8;
    size_t base = ((size_t)b * T_ + t) * C_ + h * HD_ + nl;
    attnT[base + 0]  = (_Float16)(O0[r] * inv);
    attnT[base + 16] = (_Float16)(O1[r] * inv);
    attnT[base + 32] = (_Float16)(O2[r] * inv);
    attnT[base + 48] = (_Float16)(O3[r] * inv);
  }
}

// ---------------- Kernel 5: out projection + bias + residual (f32 out) --------

__global__ void out_gemm_kernel(const _Float16* __restrict__ wgt,
                                const _Float16* __restrict__ attnT,
                                const float* __restrict__ out_b,
                                const float* __restrict__ x,
                                float* __restrict__ out) {
  __shared__ _Float16 bsm[2 * 64 * 32];
  const int b    = blockIdx.x / 32;
  const int rblk = blockIdx.x % 32;
  const int mt   = rblk >> 4;                           // 0..1
  const int nt   = rblk & 15;
  const int wave = threadIdx.x >> 5;
  const int lane = threadIdx.x & 31;
  const int m0   = mt * 256 + wave * 32;                // 0..480
  const int n0   = nt * 64;

  v8f zero = {};
  v8f acc[2][4];
#pragma unroll
  for (int i = 0; i < 2; ++i)
#pragma unroll
    for (int j = 0; j < 4; ++j) acc[i][j] = zero;

  gemm_core(wgt + (size_t)m0 * C_, attnT + ((size_t)b * T_ + n0) * C_, bsm, acc);

  const int hi = lane >> 4, nl = lane & 15;
#pragma unroll
  for (int i = 0; i < 2; ++i) {
#pragma unroll
    for (int j = 0; j < 4; ++j) {
#pragma unroll
      for (int r = 0; r < 8; ++r) {
        int o = m0 + i * 16 + r + hi * 8;
        int t = n0 + j * 16 + nl;
        size_t idx = ((size_t)b * C_ + o) * T_ + t;
        out[idx] = acc[i][j][r] + out_b[o] + x[idx];
      }
    }
  }
}

// ---------------- launch ----------------

extern "C" void kernel_launch(void* const* d_in, const int* in_sizes, int n_in,
                              void* d_out, int out_size, void* d_ws, size_t ws_size,
                              hipStream_t stream) {
  const float* x      = (const float*)d_in[0];
  const float* norm_w = (const float*)d_in[1];
  const float* norm_b = (const float*)d_in[2];
  const float* qkv_w  = (const float*)d_in[3];
  const float* qkv_b  = (const float*)d_in[4];
  const float* out_w  = (const float*)d_in[5];
  const float* out_b  = (const float*)d_in[6];
  float* out = (float*)d_out;

  char* ws = (char*)d_ws;
  size_t off = 0;
  _Float16* xnT    = (_Float16*)(ws + off); off += (size_t)B_ * T_ * C_ * 2;        // 8 MB
  _Float16* qkvw_h = (_Float16*)(ws + off); off += (size_t)3 * C_ * C_ * 2;         // 1.5 MB
  _Float16* outw_h = (_Float16*)(ws + off); off += (size_t)C_ * C_ * 2;             // 0.5 MB
  _Float16* qkvT   = (_Float16*)(ws + off); off += (size_t)B_ * T_ * 3 * C_ * 2;    // 24 MB
  _Float16* vbuf   = (_Float16*)(ws + off); off += (size_t)B_ * C_ * T_ * 2;        // 8 MB
  _Float16* attnT  = (_Float16*)(ws + off); off += (size_t)B_ * T_ * C_ * 2;        // 8 MB
  (void)off; (void)ws_size; (void)in_sizes; (void)n_in; (void)out_size;

  cvt_kernel<<<(3 * C_ * C_ + 255) / 256, 256, 0, stream>>>(qkv_w, qkvw_h, 3 * C_ * C_);
  cvt_kernel<<<(C_ * C_ + 255) / 256, 256, 0, stream>>>(out_w, outw_h, C_ * C_);
  gn_kernel<<<B_ * G_, 256, 0, stream>>>(x, norm_w, norm_b, xnT);
  qkv_gemm_kernel<<<B_ * 96, 256, 0, stream>>>(qkvw_h, xnT, qkv_b, qkvT, vbuf);
  attn_kernel<<<B_ * NH_ * (T_ / 64), 128, 0, stream>>>(qkvT, vbuf, attnT);
  out_gemm_kernel<<<B_ * 32, 256, 0, stream>>>(outw_h, attnT, out_b, x, out);
}
